// EdgeToVertexLayer_46669114638611
// MI455X (gfx1250) — matrix operand order repaired and checked by reading, not compiled, not run on hardware
//
#include <hip/hip_runtime.h>

#define D 128
#define PITCH 136   // f16 elements per LDS row: 128 + 8 pad (272B, 16B-aligned, conflict-free)

typedef __attribute__((ext_vector_type(16))) _Float16 v16h;
typedef __attribute__((ext_vector_type(8)))  _Float16 v8h;
typedef __attribute__((ext_vector_type(8)))  float    v8f;

// ---------------- kernel 1: zero the msg accumulator ----------------
__global__ void k_zero(float4* __restrict__ p, int n4) {
    int i = blockIdx.x * blockDim.x + threadIdx.x;
    if (i < n4) p[i] = make_float4(0.f, 0.f, 0.f, 0.f);
}

// ---------------- kernel 2: msg[v] += x_e[e]  (gather + atomic scatter) ---
// One wave32 per nnz row; lane handles 16 bytes (float4) -> 4 HW f32 atomics.
// Inline asm guarantees the non-returning global_atomic_add_f32 (STOREcnt,
// no CAS loop) regardless of toolchain fp-atomic policy.
__device__ __forceinline__ void hw_atomic_add_f32(float* p, float v) {
    asm volatile("global_atomic_add_f32 %0, %1, off"
                 :
                 : "v"(p), "v"(v)
                 : "memory");
}

__global__ void k_scatter(const float* __restrict__ x_e,
                          const int* __restrict__ eid,
                          const int* __restrict__ vid,
                          float* __restrict__ msg, int nnz) {
    long t   = (long)blockIdx.x * blockDim.x + threadIdx.x;
    int row  = (int)(t >> 5);
    int lane = (int)(t & 31);
    if (row >= nnz) return;
    const int e = eid[row];
    const int v = vid[row];
    const float4 val = ((const float4*)(x_e + (long)e * D))[lane];
    float* dst = msg + (long)v * D + lane * 4;
    hw_atomic_add_f32(dst + 0, val.x);
    hw_atomic_add_f32(dst + 1, val.y);
    hw_atomic_add_f32(dst + 2, val.z);
    hw_atomic_add_f32(dst + 3, val.w);
}

// ---------------- kernel 3: fused dual-GEMM + LSTM cell ----------------
// gates = msg @ W_ih^T + b_ih + h_v @ W_hh^T + b_hh, then pointwise LSTM.
// Block: 256 threads = 8 waves, M-tile = 32 vertices (2 x 16-row A tiles),
// N = 512 = 32 tiles of 16. Wave w owns N-tiles {w, w+8, w+16, w+24} so the
// i/f/g/o gate values for hidden j = w*16+col land in the same lane/slot of
// the four accumulators -> fully register-local LSTM epilogue.
__device__ __forceinline__ float sigmoidf_(float x) {
    return 1.f / (1.f + __expf(-x));
}

__global__ void __launch_bounds__(256)
k_lstm(const float* __restrict__ msg, const float* __restrict__ h_v,
       const float* __restrict__ c_v,
       const float* __restrict__ W_ih, const float* __restrict__ W_hh,
       const float* __restrict__ b_ih, const float* __restrict__ b_hh,
       float* __restrict__ h_out, float* __restrict__ c_out, int numMB) {

    __shared__ __align__(16) _Float16 sM[32 * PITCH];
    __shared__ __align__(16) _Float16 sH[32 * PITCH];

    const int tid  = threadIdx.x;
    const int w    = tid >> 5;          // wave 0..7
    const int lane = tid & 31;
    const int col  = lane & 15;         // N within tile / row within A tile
    const int j    = w * 16 + col;      // hidden index 0..127

    // per-lane gate biases (column-only -> loop invariant)
    float bias[4];
#pragma unroll
    for (int g = 0; g < 4; ++g)
        bias[g] = b_ih[g * 128 + j] + b_hh[g * 128 + j];

    // ---- preload B fragments (f32 -> f16) into registers, once ----
    // B (KxN) layout: lanes 0-15 hold K = k0..k0+15, lanes 16-31 hold K+16.
    v16h bfrag[2][4][4];                // [matrix][gate][kblock]
    {
        const float* Ws[2] = { W_ih, W_hh };
        const int kbase = (lane < 16) ? 0 : 16;
#pragma unroll
        for (int m = 0; m < 2; ++m) {
#pragma unroll
            for (int g = 0; g < 4; ++g) {
                const float* wp = Ws[m] + (long)(g * 128 + j) * D;
#pragma unroll
                for (int kb = 0; kb < 4; ++kb) {
                    const int k0 = kb * 32 + kbase;
                    float tmp[16];
                    *(float4*)(tmp + 0)  = *(const float4*)(wp + k0 + 0);
                    *(float4*)(tmp + 4)  = *(const float4*)(wp + k0 + 4);
                    *(float4*)(tmp + 8)  = *(const float4*)(wp + k0 + 8);
                    *(float4*)(tmp + 12) = *(const float4*)(wp + k0 + 12);
                    v16h f;
#pragma unroll
                    for (int i = 0; i < 16; ++i) f[i] = (_Float16)tmp[i];
                    bfrag[m][g][kb] = f;
                }
            }
        }
    }

    // staging assignments: 8 threads per vertex row, 16 f32 each (32 rows)
    const int rowT = tid >> 3;          // 0..31
    const int colT = (tid & 7) * 16;    // 0..112
    // A fragment K base: lanes 0-15 -> {0..15}, lanes 16-31 -> +16
    const int ka = (lane < 16) ? 0 : 16;

    for (int mb = blockIdx.x; mb < numMB; mb += gridDim.x) {
        const long base = (long)mb * 32 * D + (long)rowT * D + colT;

        // ---- stage msg & h_v 32x128 blocks into LDS as f16 ----
        {
            float tm[16], th[16];
#pragma unroll
            for (int q = 0; q < 4; ++q) {
                *(float4*)(tm + 4 * q) = *(const float4*)(msg + base + 4 * q);
                *(float4*)(th + 4 * q) = *(const float4*)(h_v + base + 4 * q);
            }
            v8h p0, p1, q0, q1;
#pragma unroll
            for (int i = 0; i < 8; ++i) {
                p0[i] = (_Float16)tm[i];     p1[i] = (_Float16)tm[i + 8];
                q0[i] = (_Float16)th[i];     q1[i] = (_Float16)th[i + 8];
            }
            *(v8h*)(&sM[rowT * PITCH + colT])     = p0;
            *(v8h*)(&sM[rowT * PITCH + colT + 8]) = p1;
            *(v8h*)(&sH[rowT * PITCH + colT])     = q0;
            *(v8h*)(&sH[rowT * PITCH + colT + 8]) = q1;
        }
        __syncthreads();

        // ---- load A fragments: 2 M-subtiles x 4 k-blocks, msg & h ----
        v16h aM[2][4], aH[2][4];
#pragma unroll
        for (int s = 0; s < 2; ++s) {
            const int arow = (s * 16 + col) * PITCH;
#pragma unroll
            for (int kb = 0; kb < 4; ++kb) {
                const int k0 = kb * 32 + ka;
                v8h lo = *(const v8h*)(&sM[arow + k0]);
                v8h hi = *(const v8h*)(&sM[arow + k0 + 8]);
                v16h a;
#pragma unroll
                for (int i = 0; i < 8; ++i) { a[i] = lo[i]; a[i + 8] = hi[i]; }
                aM[s][kb] = a;
                lo = *(const v8h*)(&sH[arow + k0]);
                hi = *(const v8h*)(&sH[arow + k0 + 8]);
#pragma unroll
                for (int i = 0; i < 8; ++i) { a[i] = lo[i]; a[i + 8] = hi[i]; }
                aH[s][kb] = a;
            }
        }

        // ---- 64 WMMAs: 2 M-subtiles x 4 gates x 8 k-steps ----
        v8f acc[2][4];
#pragma unroll
        for (int g = 0; g < 4; ++g) {
            v8f c0 = {}, c1 = {};
#pragma unroll
            for (int kb = 0; kb < 4; ++kb) {
                c0 = __builtin_amdgcn_wmma_f32_16x16x32_f16(
                        false, aM[0][kb], false, bfrag[0][g][kb], (short)0, c0, false, false);
                c1 = __builtin_amdgcn_wmma_f32_16x16x32_f16(
                        false, aM[1][kb], false, bfrag[0][g][kb], (short)0, c1, false, false);
            }
#pragma unroll
            for (int kb = 0; kb < 4; ++kb) {
                c0 = __builtin_amdgcn_wmma_f32_16x16x32_f16(
                        false, aH[0][kb], false, bfrag[1][g][kb], (short)0, c0, false, false);
                c1 = __builtin_amdgcn_wmma_f32_16x16x32_f16(
                        false, aH[1][kb], false, bfrag[1][g][kb], (short)0, c1, false, false);
            }
            acc[0][g] = c0;
            acc[1][g] = c1;
        }

        // ---- fused LSTM epilogue: D layout lane->N, VGPR r -> M=r(+8) ----
#pragma unroll
        for (int s = 0; s < 2; ++s) {
            const int m0 = mb * 32 + s * 16 + ((lane < 16) ? 0 : 8);
#pragma unroll
            for (int r = 0; r < 8; ++r) {
                const long off = (long)(m0 + r) * D + j;
                float iv = sigmoidf_(acc[s][0][r] + bias[0]);
                float fv = sigmoidf_(acc[s][1][r] + bias[1]);
                float gv = tanhf(acc[s][2][r] + bias[2]);
                float ov = sigmoidf_(acc[s][3][r] + bias[3]);
                float cn = fv * c_v[off] + iv * gv;
                float hn = ov * tanhf(cn);
                h_out[off] = hn;
                c_out[off] = cn;
            }
        }
        __syncthreads();   // protect LDS before next iteration's staging
    }
}

// ---------------- host launcher ----------------
extern "C" void kernel_launch(void* const* d_in, const int* in_sizes, int n_in,
                              void* d_out, int out_size, void* d_ws, size_t ws_size,
                              hipStream_t stream) {
    const float* x_e  = (const float*)d_in[0];
    const float* h_v  = (const float*)d_in[1];
    const float* c_v  = (const float*)d_in[2];
    const float* W_ih = (const float*)d_in[3];
    const float* W_hh = (const float*)d_in[4];
    const float* b_ih = (const float*)d_in[5];
    const float* b_hh = (const float*)d_in[6];
    const int*   eid  = (const int*)d_in[7];
    const int*   vid  = (const int*)d_in[8];
    // d_in[9] = v_batch: unused by the reference math

    const int V   = in_sizes[1] / D;    // 100000
    const int nnz = in_sizes[7];        // 1600000

    float* msg   = (float*)d_ws;                    // V*D f32 scratch (51.2 MB)
    float* h_out = (float*)d_out;
    float* c_out = h_out + (long)V * D;

    // 1) zero msg
    const int n4 = (V * D) / 4;
    k_zero<<<(n4 + 255) / 256, 256, 0, stream>>>((float4*)msg, n4);

    // 2) scatter-add edge features into vertices (HBM-gather bound)
    const long sthreads = (long)nnz * 32;
    k_scatter<<<(unsigned)((sthreads + 255) / 256), 256, 0, stream>>>(
        x_e, eid, vid, msg, nnz);

    // 3) fused dual-GEMM + LSTM cell (WMMA f16 -> f32), M-tile = 32
    const int numMB = V / 32;                        // 3125
    const int grid  = numMB < 1024 ? numMB : 1024;   // grid-stride amortizes B preload
    k_lstm<<<grid, 256, 0, stream>>>(msg, h_v, c_v, W_ih, W_hh, b_ih, b_hh,
                                     h_out, c_out, numMB);
}
